// HamiltonianGenerator_11922829214385
// MI455X (gfx1250) — matrix operand order repaired and checked by reading, not compiled, not run on hardware
//
#include <hip/hip_runtime.h>
#include <hip/hip_fp16.h>
#include <math.h>

typedef __attribute__((ext_vector_type(16))) _Float16 v16h;
typedef __attribute__((ext_vector_type(8)))  float    v8f;
typedef int v4i_ __attribute__((vector_size(16)));

#define BB   32      // batch
#define SS   36      // S = 2*16 + CH - 2
#define NN   256     // N
#define HH   128     // HID
#define KP   64      // K padded to multiple of 32 for the S-dim GEMMs
#define NG   144     // 4*S (LSTM gates)
#define EPSV 1e-5f

#if defined(__gfx1250__) && __has_builtin(__builtin_amdgcn_global_load_async_to_lds_b128)
#define HAVE_ASYNC_LDS 1
#endif

// Pauli 2x2 set {I, sx, "siy", sz}; KRONS[k] = kron(P[k/4], P[k%4])
__device__ __constant__ float P2c[4][2][2] = {
    {{1.f, 0.f}, {0.f, 1.f}},
    {{0.f, 1.f}, {1.f, 0.f}},
    {{0.f, 1.f}, {-1.f, 0.f}},
    {{1.f, 0.f}, {0.f, -1.f}}};

__device__ inline float sigmoidf_(float x) { return 1.f / (1.f + __expf(-x)); }

// Stage `bytes` (multiple of 16) from global to LDS, workgroup-cooperative.
// Uses CDNA5 GLOBAL_LOAD_ASYNC_TO_LDS_B128 + s_wait_asynccnt when available.
__device__ inline void stage_to_lds(const void* g, void* l, int bytes, int t,
                                    int nthr) {
#ifdef HAVE_ASYNC_LDS
  __attribute__((address_space(1))) char* gp =
      (__attribute__((address_space(1))) char*)(char*)const_cast<void*>(g);
  __attribute__((address_space(3))) char* lp =
      (__attribute__((address_space(3))) char*)l;
  for (int i = t * 16; i < bytes; i += nthr * 16)
    __builtin_amdgcn_global_load_async_to_lds_b128(
        (__attribute__((address_space(1))) v4i_*)(gp + i),
        (__attribute__((address_space(3))) v4i_*)(lp + i), 0, 0);
#if __has_builtin(__builtin_amdgcn_s_wait_asynccnt)
  __builtin_amdgcn_s_wait_asynccnt(0);
#else
  asm volatile("s_wait_asynccnt 0x0" ::: "memory");
#endif
#else
  const uint4* gp = (const uint4*)g;
  uint4* lp = (uint4*)l;
  for (int i = t; i * 16 < bytes; i += nthr) lp[i] = gp[i];
#endif
}

// ---- WMMA fragment gathers (CDNA5 16x16x32 f16 layouts, ISA 7.12.2) ----
// A: 16x32 f16, row-major source, lda in elements. Pass pointer at tile origin.
__device__ inline v16h load_a_frag_h(const _Float16* A, int lda, int lane) {
  int m  = lane & 15;
  int kh = (lane >> 4) << 3;  // lanes 16-31 hold K+8 within each 16-group
  const _Float16* row = A + m * lda;
  v16h r;
#pragma unroll
  for (int v = 0; v < 8; ++v) {
    int k = ((v >> 2) << 4) + kh + ((v & 3) << 1);
    r[2 * v]     = row[k];
    r[2 * v + 1] = row[k + 1];
  }
  return r;
}
// B: 32x16 f16 (KxN), row-major source, ldb in elements. Lane owns one column.
__device__ inline v16h load_b_frag_h(const _Float16* Bm, int ldb, int lane) {
  int n  = lane & 15;
  int kb = (lane >> 4) << 3;  // lanes 16-31 hold K+8 within each 16-group
  v16h r;
#pragma unroll
  for (int j = 0; j < 16; ++j) r[j] = Bm[(((j >> 3) << 4) + kb + (j & 7)) * ldb + n];
  return r;
}

// ---------------- prep: pack f16 weight matrices, zero K-padded acts ----------
__global__ void prep_kernel(const float* Wih, const float* bih, const float* bhh,
                            const float* mixW, const float* scWout,
                            _Float16* BIH, _Float16* BMIX, _Float16* WO16,
                            float* biasIH, _Float16* FS16, _Float16* SEQ16) {
  int tid = blockIdx.x * blockDim.x + threadIdx.x;
  int nth = gridDim.x * blockDim.x;
  // BIH (64 x 144): B[k][j] = Wih[j][k], zero-padded K
  for (int idx = tid; idx < KP * NG; idx += nth) {
    int k = idx / NG, j = idx % NG;
    BIH[idx] = (_Float16)((k < SS) ? Wih[j * SS + k] : 0.f);
  }
  // BMIX (64 x 48): B[k][o] = mix_W[o][36+k], zero-padded K and N
  for (int idx = tid; idx < KP * 48; idx += nth) {
    int k = idx / 48, o = idx % 48;
    BMIX[idx] = (_Float16)((k < SS && o < SS) ? mixW[o * 72 + SS + k] : 0.f);
  }
  // WO16: sc_Wout (36 x 128 x 256) converted to f16
  for (int idx = tid; idx < SS * HH * NN; idx += nth)
    WO16[idx] = (_Float16)scWout[idx];
  for (int j = tid; j < NG; j += nth) biasIH[j] = bih[j] + bhh[j];
  // zero full padded activation matrices (cols 36..63 must stay 0 every call)
  int tot = BB * NN * KP;
  for (int idx = tid; idx < tot; idx += nth) {
    FS16[idx] = (_Float16)0.f;
    SEQ16[idx] = (_Float16)0.f;
  }
}

// ---------------- freq / block MLPs (tiny, VALU + LDS batch-norm) -------------
__global__ void fdbd_kernel(
    const float* x, const float* fWin, const float* fbin, const float* fWmid,
    const float* fbmid, const float* fWout, const float* fbout,
    const float* fgam, const float* fbet, const float* bWin, const float* bbin,
    const float* bWmid, const float* bbmid, const float* bWout,
    const float* bbout, const float* bgam, const float* bbet, float* freqbuf,
    float* blockbuf) {
  __shared__ float xs[BB][64];
  __shared__ float ha[BB][HH];
  __shared__ float hb[BB][HH];
  const int t = threadIdx.x;  // 128 threads
  const bool isB = (blockIdx.x == 1);
  const float* Win  = isB ? bWin  : fWin;
  const float* bin  = isB ? bbin  : fbin;
  const float* Wmid = isB ? bWmid : fWmid;
  const float* bmid = isB ? bbmid : fbmid;
  const float* Wout = isB ? bWout : fWout;
  const float* bout = isB ? bbout : fbout;
  const float* gam  = isB ? bgam  : fgam;
  const float* bet  = isB ? bbet  : fbet;
  float* outbuf = isB ? blockbuf : freqbuf;
  const int xoff = isB ? 64 : 0;

  for (int idx = t; idx < BB * 64; idx += 128) {
    int b = idx >> 6, k = idx & 63;
    xs[b][k] = x[b * 128 + xoff + k];
  }
  __syncthreads();
  {
    float bias = bin[t];
    for (int b = 0; b < BB; ++b) {
      float a = bias;
      for (int k = 0; k < 64; ++k) a += xs[b][k] * Win[k * HH + t];
      ha[b][t] = fmaxf(a, 0.f);
    }
  }
  __syncthreads();
  for (int l = 0; l < 2; ++l) {
    float mu = 0.f;
    for (int b = 0; b < BB; ++b) mu += ha[b][t];
    mu *= (1.f / BB);
    float var = 0.f;
    for (int b = 0; b < BB; ++b) { float d = ha[b][t] - mu; var += d * d; }
    var *= (1.f / BB);
    float rs = rsqrtf(var + EPSV);
    float g = gam[l * HH + t], be = bet[l * HH + t];
    for (int b = 0; b < BB; ++b) hb[b][t] = g * (ha[b][t] - mu) * rs + be;
    __syncthreads();
    float bias = bmid[l * HH + t];
    const float* W = Wmid + l * HH * HH;
    for (int b = 0; b < BB; ++b) {
      float a = bias;
      for (int k = 0; k < HH; ++k) a += hb[b][k] * W[k * HH + t];
      ha[b][t] = fmaxf(a, 0.f);
    }
    __syncthreads();
  }
  {
    float mu = 0.f;
    for (int b = 0; b < BB; ++b) mu += ha[b][t];
    mu *= (1.f / BB);
    float var = 0.f;
    for (int b = 0; b < BB; ++b) { float d = ha[b][t] - mu; var += d * d; }
    var *= (1.f / BB);
    float rs = rsqrtf(var + EPSV);
    float g = gam[2 * HH + t], be = bet[2 * HH + t];
    for (int b = 0; b < BB; ++b) hb[b][t] = g * (ha[b][t] - mu) * rs + be;
  }
  __syncthreads();
  for (int idx = t; idx < BB * SS; idx += 128) {
    int b = idx / SS, s = idx % SS;
    float a = bout[s];
    for (int k = 0; k < HH; ++k) a += hb[b][k] * Wout[k * SS + s];
    outbuf[b * SS + s] = fmaxf(a, 0.f);
  }
}

// ---------------- base[b][o] = mix_b[o] + block[b] . mix_W[o, :36] ------------
__global__ void base_kernel(const float* mixW, const float* mixb,
                            const float* blockbuf, float* base) {
  int t = threadIdx.x;
  for (int idx = t; idx < BB * SS; idx += 256) {
    int b = idx / SS, o = idx % SS;
    float a = mixb[o];
    for (int c = 0; c < SS; ++c) a += mixW[o * 72 + c] * blockbuf[b * SS + c];
    base[b * SS + o] = a;
  }
}

// ---------------- per-channel sc MLPs; WMMA output layer + cos ----------------
__global__ void sc_kernel(const float* freqbuf, const float* Win,
                          const float* bin, const float* Wmid, const float* bmid,
                          const _Float16* WO16, const float* bout,
                          const float* gam, const float* bet, _Float16* FS16) {
  const int s = blockIdx.x;   // 36 channels
  const int t = threadIdx.x;  // 256 threads = 8 waves
  __shared__ float ha[BB][HH];
  __shared__ float hb[BB][HH];
  __shared__ _Float16 A16[BB][HH];
  __shared__ float xcol[BB];
  __shared__ float mus[HH], rss[HH];

  if (t < BB) xcol[t] = freqbuf[t * SS + s];  // freq.T row s == freq[:, s]
  __syncthreads();
  for (int idx = t; idx < BB * HH; idx += 256) {
    int b = idx >> 7, j = idx & 127;
    ha[b][j] = fmaxf(xcol[b] * Win[s * HH + j] + bin[s * HH + j], 0.f);
  }
  __syncthreads();
  for (int l = 0; l < 2; ++l) {
    if (t < HH) {
      float mu = 0.f;
      for (int b = 0; b < BB; ++b) mu += ha[b][t];
      mu *= (1.f / BB);
      float var = 0.f;
      for (int b = 0; b < BB; ++b) { float d = ha[b][t] - mu; var += d * d; }
      mus[t] = mu;
      rss[t] = rsqrtf(var * (1.f / BB) + EPSV);
    }
    __syncthreads();
    for (int idx = t; idx < BB * HH; idx += 256) {
      int b = idx >> 7, j = idx & 127;
      hb[b][j] = gam[s * 3 * HH + l * HH + j] * (ha[b][j] - mus[j]) * rss[j] +
                 bet[s * 3 * HH + l * HH + j];
    }
    __syncthreads();
    const float* W = Wmid + (size_t)s * 2 * HH * HH + (size_t)l * HH * HH;
    for (int idx = t; idx < BB * HH; idx += 256) {
      int b = idx >> 7, j2 = idx & 127;
      float a = bmid[s * 2 * HH + l * HH + j2];
      for (int k = 0; k < HH; ++k) a += hb[b][k] * W[k * HH + j2];
      ha[b][j2] = fmaxf(a, 0.f);
    }
    __syncthreads();
  }
  if (t < HH) {
    float mu = 0.f;
    for (int b = 0; b < BB; ++b) mu += ha[b][t];
    mu *= (1.f / BB);
    float var = 0.f;
    for (int b = 0; b < BB; ++b) { float d = ha[b][t] - mu; var += d * d; }
    mus[t] = mu;
    rss[t] = rsqrtf(var * (1.f / BB) + EPSV);
  }
  __syncthreads();
  for (int idx = t; idx < BB * HH; idx += 256) {
    int b = idx >> 7, j = idx & 127;
    A16[b][j] = (_Float16)(gam[s * 3 * HH + 2 * HH + j] * (ha[b][j] - mus[j]) * rss[j] +
                           bet[s * 3 * HH + 2 * HH + j]);
  }
  __syncthreads();
  // WMMA: (32x128) @ (128x256); 32 tiles over 8 waves, 4 K-steps each
  const int lane = t & 31, w = t >> 5;
  const _Float16* Wo = WO16 + (size_t)s * HH * NN;
  const _Float16* Ab = &A16[0][0];
  for (int tt = 0; tt < 4; ++tt) {
    int tile = w * 4 + tt;
    int Mt = tile & 1, Nt = tile >> 1;
    v8f acc = {};
    for (int k0 = 0; k0 < HH; k0 += 32) {
      v16h a  = load_a_frag_h(Ab + Mt * 16 * HH + k0, HH, lane);
      v16h bf = load_b_frag_h(Wo + (size_t)k0 * NN + Nt * 16, NN, lane);
      acc = __builtin_amdgcn_wmma_f32_16x16x32_f16(false, a, false, bf,
                                                   (short)0, acc, false, false);
    }
    int n = Nt * 16 + (lane & 15);
    float bo = bout[s * NN + n];
#pragma unroll
    for (int r = 0; r < 8; ++r) {
      int m = Mt * 16 + r + ((lane >> 4) << 3);  // batch b
      float v = fmaxf(acc[r] + bo, 0.f);
      FS16[((size_t)m * NN + n) * KP + s] = (_Float16)__cosf(v);
    }
  }
}

// ---------------- XG = freq_seq @ Wih.T + (bih+bhh): WMMA GEMM ----------------
// B matrix staged into LDS via CDNA5 async-to-LDS, shared by all 8 waves.
__global__ void xg_kernel(const _Float16* FS16, const _Float16* BIH,
                          const float* biasIH, float* XG) {
  __shared__ _Float16 Bsh[KP * NG];  // 18432 B
  stage_to_lds(BIH, Bsh, KP * NG * 2, threadIdx.x, 256);
  __syncthreads();
  const int lane = threadIdx.x & 31;
  const int w = (blockIdx.x * blockDim.x + threadIdx.x) >> 5;  // 0..4607
  const int Mt = w / 9, Nt = w % 9;  // M tiles 512, N tiles 9
  v8f acc = {};
  const _Float16* Ab = FS16 + (size_t)Mt * 16 * KP;
  for (int k0 = 0; k0 < KP; k0 += 32) {
    v16h a  = load_a_frag_h(Ab + k0, KP, lane);
    v16h bf = load_b_frag_h(Bsh + k0 * NG + Nt * 16, NG, lane);
    acc = __builtin_amdgcn_wmma_f32_16x16x32_f16(false, a, false, bf, (short)0,
                                                 acc, false, false);
  }
  int n = Nt * 16 + (lane & 15);
  float bs = biasIH[n];
#pragma unroll
  for (int r = 0; r < 8; ++r) {
    int m = Mt * 16 + r + ((lane >> 4) << 3);
    XG[(size_t)m * NG + n] = acc[r] + bs;
  }
}

// ---------------- LSTM: 32 independent recurrences, one WGP each --------------
__global__ void lstm_kernel(const float* Whh, const float* XG,
                            const float* blockbuf, _Float16* SEQ16) {
  const int b = blockIdx.x;
  const int t = threadIdx.x;  // 160 threads
  __shared__ float Wl[NG * SS];
  __shared__ float h[SS], c[SS], gates[NG];
  for (int idx = t; idx < NG * SS; idx += 160) Wl[idx] = Whh[idx];
  if (t < SS) { h[t] = blockbuf[b * SS + t]; c[t] = 0.f; }
  __syncthreads();
  for (int n = 0; n < NN; ++n) {
    if (t < NG) {
      float a = XG[((size_t)b * NN + n) * NG + t];
      const float* wr = Wl + t * SS;
      for (int k = 0; k < SS; ++k) a += wr[k] * h[k];
      gates[t] = a;
    }
    __syncthreads();
    if (t < SS) {
      float ig = sigmoidf_(gates[t]);
      float fg = sigmoidf_(gates[SS + t]);
      float gg = tanhf(gates[2 * SS + t]);
      float og = sigmoidf_(gates[3 * SS + t]);
      float cc = fg * c[t] + ig * gg;
      c[t] = cc;
      float hh = og * tanhf(cc);
      h[t] = hh;
      SEQ16[((size_t)b * NN + n) * KP + t] = (_Float16)hh;
    }
    __syncthreads();
  }
}

// ---------------- mixedT[(b,n)][o] = base[b][o] + seq[b,n,:] @ mix_W[o,36:] ---
__global__ void mix_kernel(const _Float16* SEQ16, const _Float16* BMIX,
                           const float* base, float* mixedT) {
  __shared__ _Float16 Bsh[KP * 48];  // 6144 B
  stage_to_lds(BMIX, Bsh, KP * 48 * 2, threadIdx.x, 256);
  __syncthreads();
  const int lane = threadIdx.x & 31;
  const int w = (blockIdx.x * blockDim.x + threadIdx.x) >> 5;  // 0..1535
  const int Mt = w / 3, Nt = w % 3;  // M tiles 512, N tiles 3 (48 padded cols)
  v8f acc = {};
  const _Float16* Ab = SEQ16 + (size_t)Mt * 16 * KP;
  for (int k0 = 0; k0 < KP; k0 += 32) {
    v16h a  = load_a_frag_h(Ab + k0, KP, lane);
    v16h bf = load_b_frag_h(Bsh + k0 * 48 + Nt * 16, 48, lane);
    acc = __builtin_amdgcn_wmma_f32_16x16x32_f16(false, a, false, bf, (short)0,
                                                 acc, false, false);
  }
  int o = Nt * 16 + (lane & 15);
  if (o < SS) {
#pragma unroll
    for (int r = 0; r < 8; ++r) {
      int m = Mt * 16 + r + ((lane >> 4) << 3);
      int bb = m >> 8;
      mixedT[(size_t)m * SS + o] = acc[r] + base[bb * SS + o];
    }
  }
}

// ---------------- stream 256 MB block-tridiagonal output ----------------------
__global__ void write_kernel(const float* mixedT, float* out) {
  const int bgr = blockIdx.x;  // 32*2*256 blocks
  const int r = bgr & 255, g = (bgr >> 8) & 1, b = bgr >> 9;
  const int t = threadIdx.x;  // 256 threads
  __shared__ float blk[3][16];  // [0]=sub (c=r-1), [1]=diag, [2]=super (c=r+1)
  const float* mrow = mixedT + ((size_t)b * NN + r) * SS;
  if (t < 16) {  // diagonal block: sum of 16 kron(Pauli,Pauli) terms
    int i = t >> 2, j = t & 3;
    float a = 0.f;
#pragma unroll
    for (int k = 0; k < 16; ++k)
      a += mrow[4 + g * 16 + k] * P2c[k >> 2][i >> 1][j >> 1] *
           P2c[k & 3][i & 1][j & 1];
    blk[1][t] = a;
  } else if (t < 48) {  // off-diagonals: scalar * Z1 = diag(1,1,-1,-1)
    int idx = t - 16;
    int which = idx >> 4;  // 0 = sub (channel g), 1 = super (channel 2+g)
    int e = idx & 15, i = e >> 2, j = e & 3;
    float val = mrow[which == 0 ? g : 2 + g];
    float z = (i < 2) ? 1.f : -1.f;
    blk[which == 0 ? 0 : 2][e] = (i == j) ? val * z : 0.f;
  }
  __syncthreads();
  size_t baseElem = ((size_t)(b * 2 + g) * 1024 + (size_t)4 * r) * 1024;
  float4* orow = (float4*)(out + baseElem);
  for (int q = t; q < 1024; q += 256) {  // 4 rows x 256 float4 columns
    int i = q >> 8, c4 = q & 255;
    float4 v = make_float4(0.f, 0.f, 0.f, 0.f);
    int d = c4 - r;
    if (d >= -1 && d <= 1) {
      const float* bl = blk[d + 1];
      v = make_float4(bl[i * 4 + 0], bl[i * 4 + 1], bl[i * 4 + 2], bl[i * 4 + 3]);
    }
    orow[(size_t)i * 256 + c4] = v;
  }
}

extern "C" void kernel_launch(void* const* d_in, const int* in_sizes, int n_in,
                              void* d_out, int out_size, void* d_ws,
                              size_t ws_size, hipStream_t stream) {
  (void)in_sizes; (void)n_in; (void)out_size; (void)ws_size;
  const float* x       = (const float*)d_in[0];
  const float* fd_Win  = (const float*)d_in[1];
  const float* fd_bin  = (const float*)d_in[2];
  const float* fd_Wmid = (const float*)d_in[3];
  const float* fd_bmid = (const float*)d_in[4];
  const float* fd_Wout = (const float*)d_in[5];
  const float* fd_bout = (const float*)d_in[6];
  const float* fd_gam  = (const float*)d_in[7];
  const float* fd_bet  = (const float*)d_in[8];
  const float* bd_Win  = (const float*)d_in[9];
  const float* bd_bin  = (const float*)d_in[10];
  const float* bd_Wmid = (const float*)d_in[11];
  const float* bd_bmid = (const float*)d_in[12];
  const float* bd_Wout = (const float*)d_in[13];
  const float* bd_bout = (const float*)d_in[14];
  const float* bd_gam  = (const float*)d_in[15];
  const float* bd_bet  = (const float*)d_in[16];
  const float* sc_Win  = (const float*)d_in[17];
  const float* sc_bin  = (const float*)d_in[18];
  const float* sc_Wmid = (const float*)d_in[19];
  const float* sc_bmid = (const float*)d_in[20];
  const float* sc_Wout = (const float*)d_in[21];
  const float* sc_bout = (const float*)d_in[22];
  const float* sc_gam  = (const float*)d_in[23];
  const float* sc_bet  = (const float*)d_in[24];
  const float* lstm_Wih = (const float*)d_in[25];
  const float* lstm_Whh = (const float*)d_in[26];
  const float* lstm_bih = (const float*)d_in[27];
  const float* lstm_bhh = (const float*)d_in[28];
  const float* mix_W   = (const float*)d_in[29];
  const float* mix_b   = (const float*)d_in[30];

  char* ws = (char*)d_ws;
  size_t off = 0;
  auto alloc = [&](size_t bytes) -> void* {
    off = (off + 255) & ~(size_t)255;
    void* p = ws + off;
    off += bytes;
    return p;
  };
  float*    freqbuf  = (float*)alloc(BB * SS * 4);
  float*    blockbuf = (float*)alloc(BB * SS * 4);
  float*    basep    = (float*)alloc(BB * SS * 4);
  float*    biasIH   = (float*)alloc(NG * 4);
  _Float16* BIH      = (_Float16*)alloc(KP * NG * 2);
  _Float16* BMIX     = (_Float16*)alloc(KP * 48 * 2);
  _Float16* WO16     = (_Float16*)alloc((size_t)SS * HH * NN * 2);
  _Float16* FS16     = (_Float16*)alloc((size_t)BB * NN * KP * 2);
  _Float16* SEQ16    = (_Float16*)alloc((size_t)BB * NN * KP * 2);
  float*    XG       = (float*)alloc((size_t)BB * NN * NG * 4);
  float*    mixedT   = (float*)alloc((size_t)BB * NN * SS * 4);

  prep_kernel<<<128, 256, 0, stream>>>(lstm_Wih, lstm_bih, lstm_bhh, mix_W,
                                       sc_Wout, BIH, BMIX, WO16, biasIH, FS16,
                                       SEQ16);
  fdbd_kernel<<<2, 128, 0, stream>>>(
      x, fd_Win, fd_bin, fd_Wmid, fd_bmid, fd_Wout, fd_bout, fd_gam, fd_bet,
      bd_Win, bd_bin, bd_Wmid, bd_bmid, bd_Wout, bd_bout, bd_gam, bd_bet,
      freqbuf, blockbuf);
  base_kernel<<<1, 256, 0, stream>>>(mix_W, mix_b, blockbuf, basep);
  sc_kernel<<<SS, 256, 0, stream>>>(freqbuf, sc_Win, sc_bin, sc_Wmid, sc_bmid,
                                    WO16, sc_bout, sc_gam, sc_bet, FS16);
  xg_kernel<<<576, 256, 0, stream>>>(FS16, BIH, biasIH, XG);
  lstm_kernel<<<BB, 160, 0, stream>>>(lstm_Whh, XG, blockbuf, SEQ16);
  mix_kernel<<<192, 256, 0, stream>>>(SEQ16, BMIX, basep, mixedT);
  write_kernel<<<BB * 2 * NN, 256, 0, stream>>>(mixedT, (float*)d_out);
}
